// FMoE_36472862277759
// MI455X (gfx1250) — compile-verified, matrix operand order checked by reading, not run on hardware
//
#include <hip/hip_runtime.h>
#include <math.h>

#define T_TOK 4096
#define DIM   1024
#define HID   1024
#define NE    16
#define CAPC  1024

// LDS strides (padded, 16B-aligned rows)
#define SA   40    // A row stride in bf16 elements (80 B)
#define SBP  20    // packed-B column stride in uints (80 B), fragment-ordered

typedef __attribute__((ext_vector_type(16))) __bf16 v16bf;
typedef __attribute__((ext_vector_type(8)))  float  v8f;

__device__ __forceinline__ unsigned short f2bf(float f) {
  unsigned u = __float_as_uint(f);
  u += 0x7FFFu + ((u >> 16) & 1u);            // round-to-nearest-even
  return (unsigned short)(u >> 16);
}
__device__ __forceinline__ unsigned pack2(float lo, float hi) {
  return (unsigned)f2bf(lo) | ((unsigned)f2bf(hi) << 16);
}

// Fragment-order permutation of k-pair index p (0..15) so that a lane's
// 8 B-fragment uints are contiguous:  hi=0 needs p{0-3,8-11} -> q 0..7,
// hi=1 needs p{4-7,12-15} -> q 8..15.
__device__ __forceinline__ int qperm(int p) {
  return (p & 3) | ((p & 8) >> 1) | ((p & 4) << 1);
}

// A fragment: lane holds row m; VGPR i holds K pair kb,kb+1 with
// kb = (i<4?0:16) + hi*8 + (i&3)*2  -> two contiguous 16B runs per lane.
__device__ __forceinline__ v16bf load_a_frag(const unsigned short* As, int row, int hi) {
  union { v16bf v; uint4 q[2]; } r;
  const unsigned short* base = As + row * SA + hi * 8;
  r.q[0] = *(const uint4*)(base);        // K = hi*8 .. hi*8+7
  r.q[1] = *(const uint4*)(base + 16);   // K = 16+hi*8 .. 16+hi*8+7
  return r.v;
}
// B fragment: lane holds column n; column-major fragment-ordered store makes
// this two contiguous aligned b128 loads.
__device__ __forceinline__ v16bf load_b_frag(const unsigned* BsP, int col, int hi) {
  union { v16bf v; uint4 q[2]; } r;
  const unsigned* base = BsP + col * SBP + hi * 8;
  r.q[0] = *(const uint4*)(base);
  r.q[1] = *(const uint4*)(base + 4);
  return r.v;
}

// Shared 128x128-tile main loop over K=1024 in steps of 32.
// Ag: bf16 activations, row stride 1024. Bg: fp32 weights, row stride 1024.
// A tile is staged global->LDS with async-to-LDS (ASYNCcnt); B tile goes
// through VGPRs because it converts fp32 -> packed bf16 pairs in flight.
__device__ __forceinline__ void wmma_mainloop(const unsigned short* Ag, const float* Bg,
                                              unsigned short* As, unsigned* BsP,
                                              int m0, int n0, int wm, int wn,
                                              int lane, int tid, v8f acc[2][4]) {
  const int hi = lane >> 4;
  const int lr = lane & 15;
  // Low 32 bits of a generic shared pointer = LDS offset (flat aperture maps
  // addr[31:0] onto the wave's LDS allocation, matching async LDS addressing).
  const unsigned lds_as_base = (unsigned)(size_t)(const void*)As;
  for (int k0 = 0; k0 < 1024; k0 += 32) {
    __syncthreads();
    // Stage A: 128 rows x 32 bf16, async global->LDS, 16B per lane per issue.
#pragma unroll
    for (int j = 0; j < 2; ++j) {
      int c = tid + j * 256;
      int row = c >> 2, seg = c & 3;
      unsigned ldst = lds_as_base + (unsigned)((row * SA + seg * 8) * 2);
      const unsigned short* gsrc = Ag + (size_t)(m0 + row) * 1024 + k0 + seg * 8;
      asm volatile("global_load_async_to_lds_b128 %0, %1, off"
                   :: "v"(ldst), "v"(gsrc)
                   : "memory");
    }
    // Stage B: 32 k-rows x 128 cols fp32 -> bf16 k-pairs, stored column-major
    // in fragment order (see qperm).
#pragma unroll
    for (int j = 0; j < 2; ++j) {
      int c = tid + j * 256;
      int p = c >> 5, nseg = c & 31;
      const float* r0 = Bg + (size_t)(k0 + 2 * p) * 1024 + n0 + nseg * 4;
      float4 f0 = *(const float4*)r0;
      float4 f1 = *(const float4*)(r0 + 1024);
      unsigned* colbase = BsP + nseg * 4 * SBP + qperm(p);
      colbase[0 * SBP] = pack2(f0.x, f1.x);
      colbase[1 * SBP] = pack2(f0.y, f1.y);
      colbase[2 * SBP] = pack2(f0.z, f1.z);
      colbase[3 * SBP] = pack2(f0.w, f1.w);
    }
    // Prefetch next K-step of the weight tile into cache (global_prefetch_b8).
    if (k0 + 32 < 1024) {
      __builtin_prefetch(Bg + (size_t)(k0 + 32 + 2 * (tid >> 5)) * 1024 + n0 + (tid & 31) * 4,
                         0, 1);
    }
    // Our wave's async copies must land before the barrier publishes the tile.
    asm volatile("s_wait_asynccnt 0" ::: "memory");
    __syncthreads();
    v16bf a0 = load_a_frag(As, wm * 32 + lr,      hi);
    v16bf a1 = load_a_frag(As, wm * 32 + 16 + lr, hi);
#pragma unroll
    for (int nt = 0; nt < 4; ++nt) {
      v16bf b = load_b_frag(BsP, wn * 64 + nt * 16 + lr, hi);
      acc[0][nt] = __builtin_amdgcn_wmma_f32_16x16x32_bf16(
          false, a0, false, b, (short)0, acc[0][nt], false, false);
      acc[1][nt] = __builtin_amdgcn_wmma_f32_16x16x32_bf16(
          false, a1, false, b, (short)0, acc[1][nt], false, false);
    }
  }
}

// ---------------- init: zero out, reset slots/counts ----------------
__global__ __launch_bounds__(256) void moe_init(float* out, int* slot_token,
                                                float* slot_weight, int* counts) {
  int i = blockIdx.x * 256 + threadIdx.x;
  if (i < T_TOK * DIM) out[i] = 0.0f;
  if (i < NE * CAPC) { slot_token[i] = -1; slot_weight[i] = 0.0f; }
  if (i < NE) counts[i] = 0;
}

// ---------------- gate + dispatch (one wave32 per token) ----------------
__global__ __launch_bounds__(256) void moe_gate_dispatch(const float* __restrict__ x,
                                                         const float* __restrict__ Wg,
                                                         const float* __restrict__ bg,
                                                         int* counts, int* slot_token,
                                                         float* slot_weight,
                                                         unsigned short* buf) {
  const int lane = threadIdx.x & 31;
  const int wave = threadIdx.x >> 5;
  const int t = blockIdx.x * 8 + wave;
  const float* xt = x + (size_t)t * DIM;

  float acc[NE];
#pragma unroll
  for (int e = 0; e < NE; ++e) acc[e] = 0.0f;
  for (int d = lane; d < DIM; d += 32) {
    float xv = xt[d];
    const float* wr = Wg + d * NE;
#pragma unroll
    for (int e = 0; e < NE; ++e) acc[e] += xv * wr[e];
  }
#pragma unroll
  for (int e = 0; e < NE; ++e) {
    float v = acc[e];
#pragma unroll
    for (int off = 16; off > 0; off >>= 1) v += __shfl_down(v, off, 32);
    acc[e] = v;   // valid on lane 0
  }

  int e0 = 0, e1 = 0, pos0 = CAPC, pos1 = CAPC;
  if (lane == 0) {
    float v0 = -3.0e38f, v1 = -3.0e38f;
    int b0 = 0, b1i = 0;
#pragma unroll
    for (int e = 0; e < NE; ++e) {
      float le = acc[e] + bg[e];
      if (le > v0)       { v1 = v0; b1i = b0; v0 = le; b0 = e; }
      else if (le > v1)  { v1 = le; b1i = e; }
    }
    float ew = __expf(v1 - v0);           // softmax over the top-2 logits
    float w0 = 1.0f / (1.0f + ew);
    float w1 = 1.0f - w0;
    e0 = b0; e1 = b1i;
    pos0 = atomicAdd(counts + e0, 1);
    pos1 = atomicAdd(counts + e1, 1);
    if (pos0 < CAPC) { slot_token[e0 * CAPC + pos0] = t; slot_weight[e0 * CAPC + pos0] = w0; }
    if (pos1 < CAPC) { slot_token[e1 * CAPC + pos1] = t; slot_weight[e1 * CAPC + pos1] = w1; }
  }
  e0 = __shfl(e0, 0, 32); pos0 = __shfl(pos0, 0, 32);
  e1 = __shfl(e1, 0, 32); pos1 = __shfl(pos1, 0, 32);

  if (pos0 < CAPC) {
    unsigned short* dst = buf + ((size_t)e0 * CAPC + pos0) * DIM;
    for (int d = lane; d < DIM; d += 32) dst[d] = f2bf(xt[d]);
  }
  if (pos1 < CAPC) {
    unsigned short* dst = buf + ((size_t)e1 * CAPC + pos1) * DIM;
    for (int d = lane; d < DIM; d += 32) dst[d] = f2bf(xt[d]);
  }
}

// ---------------- GEMM1: h = relu(buf @ W1 + b1) -> bf16 ----------------
__global__ __launch_bounds__(256) void moe_gemm1(const unsigned short* __restrict__ buf,
                                                 const float* __restrict__ W1,
                                                 const float* __restrict__ b1,
                                                 unsigned short* __restrict__ h) {
  __shared__ unsigned short As[128 * SA];
  __shared__ unsigned BsP[128 * SBP];
  const int tid = threadIdx.x;
  const int lane = tid & 31, wave = tid >> 5;
  const int wm = wave & 3, wn = wave >> 2;
  const int e  = blockIdx.z;
  const int m0 = blockIdx.y * 128, n0 = blockIdx.x * 128;

  v8f zero = {0.f, 0.f, 0.f, 0.f, 0.f, 0.f, 0.f, 0.f};
  v8f acc[2][4];
#pragma unroll
  for (int i = 0; i < 2; ++i)
#pragma unroll
    for (int j = 0; j < 4; ++j) acc[i][j] = zero;

  wmma_mainloop(buf + (size_t)e * CAPC * 1024, W1 + (size_t)e * 1024 * 1024,
                As, BsP, m0, n0, wm, wn, lane, tid, acc);

  const float* b1e = b1 + e * HID;
  unsigned short* he = h + (size_t)e * CAPC * HID;
  const int hi = lane >> 4, nc = lane & 15;
#pragma unroll
  for (int mt = 0; mt < 2; ++mt)
#pragma unroll
    for (int r = 0; r < 8; ++r) {
      int m = m0 + wm * 32 + mt * 16 + hi * 8 + r;
#pragma unroll
      for (int nt = 0; nt < 4; ++nt) {
        int n = n0 + wn * 64 + nt * 16 + nc;
        float v = acc[mt][nt][r] + b1e[n];
        he[(size_t)m * HID + n] = f2bf(v > 0.0f ? v : 0.0f);
      }
    }
}

// ------- GEMM2 + fused combine: out[t] += w * (h @ W2 + b2) -------
__global__ __launch_bounds__(256) void moe_gemm2(const unsigned short* __restrict__ h,
                                                 const float* __restrict__ W2,
                                                 const float* __restrict__ b2,
                                                 const int* __restrict__ slot_token,
                                                 const float* __restrict__ slot_weight,
                                                 float* out) {
  __shared__ unsigned short As[128 * SA];
  __shared__ unsigned BsP[128 * SBP];
  const int tid = threadIdx.x;
  const int lane = tid & 31, wave = tid >> 5;
  const int wm = wave & 3, wn = wave >> 2;
  const int e  = blockIdx.z;
  const int m0 = blockIdx.y * 128, n0 = blockIdx.x * 128;

  v8f zero = {0.f, 0.f, 0.f, 0.f, 0.f, 0.f, 0.f, 0.f};
  v8f acc[2][4];
#pragma unroll
  for (int i = 0; i < 2; ++i)
#pragma unroll
    for (int j = 0; j < 4; ++j) acc[i][j] = zero;

  wmma_mainloop(h + (size_t)e * CAPC * 1024, W2 + (size_t)e * 1024 * 1024,
                As, BsP, m0, n0, wm, wn, lane, tid, acc);

  const float* b2e = b2 + e * DIM;
  const int hi = lane >> 4, nc = lane & 15;
#pragma unroll
  for (int mt = 0; mt < 2; ++mt)
#pragma unroll
    for (int r = 0; r < 8; ++r) {
      int m = m0 + wm * 32 + mt * 16 + hi * 8 + r;
      int t = slot_token[e * CAPC + m];
      if (t < 0) continue;                         // unused capacity slot
      float wv = slot_weight[e * CAPC + m];
      float* od = out + (size_t)t * DIM;
#pragma unroll
      for (int nt = 0; nt < 4; ++nt) {
        int n = n0 + wn * 64 + nt * 16 + nc;
        atomicAdd(od + n, wv * (acc[mt][nt][r] + b2e[n]));
      }
    }
}

extern "C" void kernel_launch(void* const* d_in, const int* in_sizes, int n_in,
                              void* d_out, int out_size, void* d_ws, size_t ws_size,
                              hipStream_t stream) {
  (void)in_sizes; (void)n_in; (void)out_size; (void)ws_size;
  const float* x  = (const float*)d_in[0];
  const float* Wg = (const float*)d_in[1];
  const float* bg = (const float*)d_in[2];
  const float* W1 = (const float*)d_in[3];
  const float* b1 = (const float*)d_in[4];
  const float* W2 = (const float*)d_in[5];
  const float* b2 = (const float*)d_in[6];
  float* out = (float*)d_out;

  char* ws = (char*)d_ws;
  int*   counts      = (int*)(ws + 0);
  int*   slot_token  = (int*)(ws + 256);
  float* slot_weight = (float*)(ws + 256 + (size_t)NE * CAPC * 4);
  unsigned short* buf = (unsigned short*)(ws + 131328);                     // bf16 [E][CAP][D]
  unsigned short* hbf = (unsigned short*)(ws + 131328 + (size_t)33554432);  // bf16 [E][CAP][H]

  moe_init<<<(T_TOK * DIM + 255) / 256, 256, 0, stream>>>(out, slot_token, slot_weight, counts);
  moe_gate_dispatch<<<T_TOK / 8, 256, 0, stream>>>(x, Wg, bg, counts, slot_token,
                                                   slot_weight, buf);
  dim3 gg(HID / 128, CAPC / 128, NE);
  moe_gemm1<<<gg, 256, 0, stream>>>(buf, W1, b1, hbf);
  dim3 g2(DIM / 128, CAPC / 128, NE);
  moe_gemm2<<<g2, 256, 0, stream>>>(hbf, W2, b2, slot_token, slot_weight, out);
}
